// HierarchicalDecoder_5480378270057
// MI455X (gfx1250) — matrix operand order repaired and checked by reading, not compiled, not run on hardware
//
#include <hip/hip_runtime.h>

// ---------------- sizes ----------------
constexpr int BATCH  = 128;
constexpr int HID    = 512;
constexpr int NOUT   = 176;
constexpr int TSTEPS = 256;    // TC*TD
constexpr int NGATE  = 2048;   // 4*H

// GEMM1 combined K: [note 176 -> pad 192][h1 512] = 704 -> 22 ktiles of 32
constexpr int NK1 = 22;
// GEMM2 combined K: [h1 512][h2 512] = 1024 -> 32 ktiles
constexpr int NK2 = 32;
// GEMM3 K: h2 512 -> 16 ktiles ; N = 176 -> 11 ntiles
constexpr int NKP = 16;
constexpr int NTP = 11;

// LDS activation buffer strides (elements). Chosen so the dword stride per
// row is odd -> conflict-free across the 16 rows read by the A-fragment.
constexpr int X1W = 194;   // note buffer: 192 cols + 2 pad -> 97 dwords/row
constexpr int X2W = 1026;  // [h1(512) | h2(512)] + 2 pad -> 513 dwords/row

// per (ntile,ktile) B-fragment block: 32 lanes x 16 bf16 = 1024 B
constexpr int BFRAG_U4 = 64;   // in uint4 units

typedef __attribute__((ext_vector_type(16))) __bf16 v16bf;
typedef __attribute__((ext_vector_type(8)))  float  v8f;

union BF16x16 {
  v16bf v;
  unsigned int u[8];
  uint4 q[2];
};

__device__ __forceinline__ unsigned short f2bf(float f) {
  unsigned int u = __float_as_uint(f);
  u += 0x7FFFu + ((u >> 16) & 1u);   // round-to-nearest-even
  return (unsigned short)(u >> 16);
}
__device__ __forceinline__ float sigf(float x) { return 1.0f / (1.0f + __expf(-x)); }

__device__ __forceinline__ v8f wmma_bf16(v16bf a, v16bf b, v8f c) {
  return __builtin_amdgcn_wmma_f32_16x16x32_bf16(false, a, false, b, (short)0, c,
                                                 false, false);
}

// B-fragment from a pre-swizzled block pointer (32 contiguous bytes per lane)
__device__ __forceinline__ v16bf load_bfrag_p(const uint4* __restrict__ p) {
  BF16x16 b;
  b.q[0] = p[0];
  b.q[1] = p[1];
  return b.v;
}

// A-fragment (16x32 M x K) from a row-major bf16 LDS row `xs` (dword view of
// row M = lane&15). kbase already includes ktile*32 + 8*(lane>=16).
__device__ __forceinline__ v16bf load_afrag(const unsigned int* __restrict__ xs,
                                            int kbase) {
  BF16x16 a;
#pragma unroll
  for (int j = 0; j < 8; ++j) {
    int kk = kbase + ((j < 4) ? 2 * j : 16 + 2 * (j - 4));  // K pair index (even)
    a.u[j] = xs[kk >> 1];
  }
  return a.v;
}

// =====================================================================
// Main persistent decoder: 8 workgroups x 1024 threads (32 waves).
// Workgroup b owns batch rows [16b, 16b+16); no inter-WG sync needed.
// Wave w owns h-columns [16w, 16w+16): it computes the four gate tiles
// (i,f,g,o at N = 16w + {0,512,1024,1536}) and does the LSTM pointwise
// with c-state held in fp32 registers.
// =====================================================================
__global__ __launch_bounds__(1024) void hd_decoder_kernel(
    const unsigned short* __restrict__ Wc1s,  // swizzled [note|U1] K=704,N=2048
    const unsigned short* __restrict__ Wc2s,  // swizzled [W2|U2]  K=1024,N=2048
    const unsigned short* __restrict__ Wps,   // swizzled Wp       K=512,N=176
    const float* __restrict__ g1c,            // [128][2048] cond part of g1 (+b1)
    const unsigned short* __restrict__ dh1,   // [128][512] bf16 segment-reset h1
    const unsigned short* __restrict__ dh2,
    const float* __restrict__ dc1,            // [128][512] f32 segment-reset c1
    const float* __restrict__ dc2,
    const float* __restrict__ b2,             // dec_b2 [2048]
    const float* __restrict__ bpv,            // bp [176]
    float* __restrict__ out)                  // [128][256][176]
{
  __shared__ unsigned short x1buf[16][X1W];   // note (cols 0..175, rest zero)
  __shared__ unsigned short x2buf[16][X2W];   // [h1 | h2] bf16
  __shared__ float logits[16][NOUT];

  const int tid  = threadIdx.x;
  const int lane = tid & 31;
  const int wave = tid >> 5;
  const int nlo  = lane & 15;
  const int half8 = (lane >> 4) * 8;          // 0 or 8: M-offset AND A-K-offset
  const int rowbase = blockIdx.x * 16;
  const int colw = wave * 16 + nlo;           // h-column owned in pointwise

  const unsigned int* xs1 = (const unsigned int*)&x1buf[nlo][0];
  const unsigned int* xs2 = (const unsigned int*)&x2buf[nlo][0];

  // per-wave swizzled weight base pointers (gate tiles i,f,g,o)
  const uint4* W1i0 = (const uint4*)(Wc1s + (((size_t)(wave)      * NK1) * 32 + lane) * 16);
  const uint4* W1f0 = (const uint4*)(Wc1s + (((size_t)(wave + 32) * NK1) * 32 + lane) * 16);
  const uint4* W1g0 = (const uint4*)(Wc1s + (((size_t)(wave + 64) * NK1) * 32 + lane) * 16);
  const uint4* W1o0 = (const uint4*)(Wc1s + (((size_t)(wave + 96) * NK1) * 32 + lane) * 16);
  const uint4* W2i0 = (const uint4*)(Wc2s + (((size_t)(wave)      * NK2) * 32 + lane) * 16);
  const uint4* W2f0 = (const uint4*)(Wc2s + (((size_t)(wave + 32) * NK2) * 32 + lane) * 16);
  const uint4* W2g0 = (const uint4*)(Wc2s + (((size_t)(wave + 64) * NK2) * 32 + lane) * 16);
  const uint4* W2o0 = (const uint4*)(Wc2s + (((size_t)(wave + 96) * NK2) * 32 + lane) * 16);
  const uint4* Wp0  = (const uint4*)(Wps  + (((size_t)wave * NKP) * 32 + lane) * 16);

  float c1r[8], c2r[8];

  // zero note buffer (incl. padding) once; note carries across all 256 steps
  for (int idx = tid; idx < 16 * X1W; idx += 1024)
    ((unsigned short*)x1buf)[idx] = 0;

  for (int t = 0; t < TSTEPS; ++t) {
    if ((t & 15) == 0) {
      // segment start: reset h/c to decoder-initial states
#pragma unroll
      for (int j = 0; j < 8; ++j) {
        const int rg = (rowbase + half8 + j) * HID + colw;
        c1r[j] = dc1[rg];
        c2r[j] = dc2[rg];
        x2buf[half8 + j][colw]       = dh1[rg];
        x2buf[half8 + j][512 + colw] = dh2[rg];
      }
    }
    __syncthreads();

    // ---------------- Layer-1 GEMM: g1 = [note|h1] @ [W1note|U1] + g1_const
    v8f ai, af, ag, ao;
    {
      const float* gc = g1c + (size_t)rowbase * NGATE;
#pragma unroll
      for (int j = 0; j < 8; ++j) {
        const float* r = gc + (size_t)(half8 + j) * NGATE;
        ai[j] = r[colw];
        af[j] = r[colw + 512];
        ag[j] = r[colw + 1024];
        ao[j] = r[colw + 1536];
      }
    }
    {
      const uint4* pi = W1i0;
      const uint4* pf = W1f0;
      const uint4* pg = W1g0;
      const uint4* po = W1o0;
      // note part: ktiles 0..5 read from x1buf
#pragma clang loop unroll(disable)
      for (int kt = 0; kt < 6; ++kt) {
        v16bf bi_ = load_bfrag_p(pi); pi += BFRAG_U4;
        v16bf bf_ = load_bfrag_p(pf); pf += BFRAG_U4;
        v16bf bg_ = load_bfrag_p(pg); pg += BFRAG_U4;
        v16bf bo_ = load_bfrag_p(po); po += BFRAG_U4;
        v16bf a = load_afrag(xs1, kt * 32 + half8);
        ai = wmma_bf16(a, bi_, ai);
        af = wmma_bf16(a, bf_, af);
        ag = wmma_bf16(a, bg_, ag);
        ao = wmma_bf16(a, bo_, ao);
      }
      // h1 part: ktiles 6..21 read from x2buf[0:512]
#pragma clang loop unroll(disable)
      for (int kt = 0; kt < 16; ++kt) {
        v16bf bi_ = load_bfrag_p(pi); pi += BFRAG_U4;
        v16bf bf_ = load_bfrag_p(pf); pf += BFRAG_U4;
        v16bf bg_ = load_bfrag_p(pg); pg += BFRAG_U4;
        v16bf bo_ = load_bfrag_p(po); po += BFRAG_U4;
        v16bf a = load_afrag(xs2, kt * 32 + half8);
        ai = wmma_bf16(a, bi_, ai);
        af = wmma_bf16(a, bf_, af);
        ag = wmma_bf16(a, bg_, ag);
        ao = wmma_bf16(a, bo_, ao);
      }
    }
    __syncthreads();   // all GEMM1 reads of x2buf[h1] done before we overwrite

    // ---------------- Layer-1 pointwise (c in registers), h1 -> LDS bf16
#pragma unroll
    for (int j = 0; j < 8; ++j) {
      float c = sigf(af[j]) * c1r[j] + sigf(ai[j]) * tanhf(ag[j]);
      c1r[j] = c;
      x2buf[half8 + j][colw] = f2bf(sigf(ao[j]) * tanhf(c));
    }
    __syncthreads();

    // ---------------- Layer-2 GEMM: g2 = [h1|h2] @ [W2|U2] + b2
    {
      float bi = b2[colw], bf_ = b2[colw + 512];
      float bg = b2[colw + 1024], bo = b2[colw + 1536];
#pragma unroll
      for (int j = 0; j < 8; ++j) { ai[j] = bi; af[j] = bf_; ag[j] = bg; ao[j] = bo; }
    }
    {
      const uint4* pi = W2i0;
      const uint4* pf = W2f0;
      const uint4* pg = W2g0;
      const uint4* po = W2o0;
#pragma clang loop unroll(disable)
      for (int kt = 0; kt < NK2; ++kt) {
        v16bf bi_ = load_bfrag_p(pi); pi += BFRAG_U4;
        v16bf bf_ = load_bfrag_p(pf); pf += BFRAG_U4;
        v16bf bg_ = load_bfrag_p(pg); pg += BFRAG_U4;
        v16bf bo_ = load_bfrag_p(po); po += BFRAG_U4;
        v16bf a = load_afrag(xs2, kt * 32 + half8);
        ai = wmma_bf16(a, bi_, ai);
        af = wmma_bf16(a, bf_, af);
        ag = wmma_bf16(a, bg_, ag);
        ao = wmma_bf16(a, bo_, ao);
      }
    }
    __syncthreads();   // GEMM2 reads of x2buf[h2] done before overwrite

    // ---------------- Layer-2 pointwise, h2 -> LDS bf16
#pragma unroll
    for (int j = 0; j < 8; ++j) {
      float c = sigf(af[j]) * c2r[j] + sigf(ai[j]) * tanhf(ag[j]);
      c2r[j] = c;
      x2buf[half8 + j][512 + colw] = f2bf(sigf(ao[j]) * tanhf(c));
    }
    __syncthreads();

    // ---------------- Output projection: logits = h2 @ Wp + bp (11 ntiles)
    if (wave < NTP) {
      v8f ap;
      float bb = bpv[wave * 16 + nlo];
#pragma unroll
      for (int j = 0; j < 8; ++j) ap[j] = bb;
      const uint4* pp = Wp0;
#pragma clang loop unroll(disable)
      for (int kt = 0; kt < NKP; ++kt) {
        v16bf bb_ = load_bfrag_p(pp); pp += BFRAG_U4;
        v16bf a = load_afrag(xs2, 512 + kt * 32 + half8);
        ap = wmma_bf16(a, bb_, ap);
      }
#pragma unroll
      for (int j = 0; j < 8; ++j) logits[half8 + j][wave * 16 + nlo] = ap[j];
    }
    __syncthreads();

    // ---------------- row softmax (16 rows x 32 lanes), note -> LDS + out
    if (tid < 512) {
      const int row = tid >> 5;
      const int l   = tid & 31;
      float v[6];
      float m = -3.4e38f;
#pragma unroll
      for (int ii = 0; ii < 6; ++ii) {
        int col = l + 32 * ii;
        v[ii] = (col < NOUT) ? logits[row][col] : -3.4e38f;
        m = fmaxf(m, v[ii]);
      }
#pragma unroll
      for (int off = 16; off > 0; off >>= 1) m = fmaxf(m, __shfl_xor(m, off, 32));
      float s = 0.f;
#pragma unroll
      for (int ii = 0; ii < 6; ++ii) {
        int col = l + 32 * ii;
        if (col < NOUT) { v[ii] = __expf(v[ii] - m); s += v[ii]; }
      }
#pragma unroll
      for (int off = 16; off > 0; off >>= 1) s += __shfl_xor(s, off, 32);
      float inv = 1.0f / s;
      float* orow = out + ((size_t)(rowbase + row) * TSTEPS + t) * NOUT;
#pragma unroll
      for (int ii = 0; ii < 6; ++ii) {
        int col = l + 32 * ii;
        if (col < NOUT) {
          float p = v[ii] * inv;
          x1buf[row][col] = f2bf(p);
          orow[col] = p;
        }
      }
    }
  }
}

// =====================================================================
// One-time prologue kernels (tiny fraction of total work -> simple fp32)
// =====================================================================

// Y[b][n] = tanh(X[b,:K] @ W[:,n] + bias[n]),  W row-major K x N
__global__ void hd_dense_tanh(const float* __restrict__ X, const float* __restrict__ W,
                              const float* __restrict__ bias, float* __restrict__ Y,
                              int K, int N) {
  int idx = blockIdx.x * blockDim.x + threadIdx.x;
  if (idx >= BATCH * N) return;
  int b = idx / N, n = idx - b * N;
  const float* x = X + (size_t)b * K;
  float acc = bias[n];
  for (int k = 0; k < K; ++k) acc += x[k] * W[(size_t)k * N + n];
  Y[idx] = tanhf(acc);
}

// conductor layer 1 (input x0 == 0 so W1 drops out)
__global__ void hd_cond1(const float* __restrict__ s, const float* __restrict__ U1,
                         const float* __restrict__ b1, float* __restrict__ h1c) {
  int idx = blockIdx.x * blockDim.x + threadIdx.x;
  if (idx >= BATCH * HID) return;
  int b = idx / HID, n = idx - b * HID;
  const float* ch1 = s + (size_t)b * NGATE;
  const float* cc1 = ch1 + 512;
  float gi = b1[n], gf = b1[n + 512], gg = b1[n + 1024], go = b1[n + 1536];
  for (int k = 0; k < HID; ++k) {
    float hv = ch1[k];
    const float* w = U1 + (size_t)k * NGATE + n;
    gi += hv * w[0]; gf += hv * w[512]; gg += hv * w[1024]; go += hv * w[1536];
  }
  float c = sigf(gf) * cc1[n] + sigf(gi) * tanhf(gg);
  h1c[idx] = sigf(go) * tanhf(c);
}

// conductor layer 2 -> cond_emb
__global__ void hd_cond2(const float* __restrict__ s, const float* __restrict__ h1c,
                         const float* __restrict__ W2, const float* __restrict__ U2,
                         const float* __restrict__ b2c, float* __restrict__ ce) {
  int idx = blockIdx.x * blockDim.x + threadIdx.x;
  if (idx >= BATCH * HID) return;
  int b = idx / HID, n = idx - b * HID;
  const float* ch2 = s + (size_t)b * NGATE + 1024;
  const float* cc2 = ch2 + 512;
  const float* x = h1c + (size_t)b * HID;
  float gi = b2c[n], gf = b2c[n + 512], gg = b2c[n + 1024], go = b2c[n + 1536];
  for (int k = 0; k < HID; ++k) {
    float xv = x[k], hv = ch2[k];
    const float* w = W2 + (size_t)k * NGATE + n;
    const float* u = U2 + (size_t)k * NGATE + n;
    gi += xv * w[0] + hv * u[0];
    gf += xv * w[512] + hv * u[512];
    gg += xv * w[1024] + hv * u[1024];
    go += xv * w[1536] + hv * u[1536];
  }
  float c = sigf(gf) * cc2[n] + sigf(gi) * tanhf(gg);
  ce[idx] = sigf(go) * tanhf(c);
}

// sd = tanh(ce @ Wcd + bcd) -> split into decoder initial states
__global__ void hd_decinit(const float* __restrict__ ce, const float* __restrict__ Wcd,
                           const float* __restrict__ bcd,
                           unsigned short* __restrict__ dh1, float* __restrict__ dc1,
                           unsigned short* __restrict__ dh2, float* __restrict__ dc2) {
  int idx = blockIdx.x * blockDim.x + threadIdx.x;
  if (idx >= BATCH * NGATE) return;
  int b = idx / NGATE, n = idx - b * NGATE;
  const float* x = ce + (size_t)b * HID;
  float acc = bcd[n];
  for (int k = 0; k < HID; ++k) acc += x[k] * Wcd[(size_t)k * NGATE + n];
  float v = tanhf(acc);
  int q = n >> 9, c = n & 511;
  if (q == 0)      dh1[b * HID + c] = f2bf(v);
  else if (q == 1) dc1[b * HID + c] = v;
  else if (q == 2) dh2[b * HID + c] = f2bf(v);
  else             dc2[b * HID + c] = v;
}

// g1_const = ce @ dec_W1[176:688,:] + dec_b1  (constant part of layer-1 gates)
__global__ void hd_g1const(const float* __restrict__ ce, const float* __restrict__ dW1,
                           const float* __restrict__ db1, float* __restrict__ g1c) {
  int idx = blockIdx.x * blockDim.x + threadIdx.x;
  if (idx >= BATCH * NGATE) return;
  int b = idx / NGATE, n = idx - b * NGATE;
  const float* x = ce + (size_t)b * HID;
  float acc = db1[n];
  for (int k = 0; k < HID; ++k) acc += x[k] * dW1[(size_t)(176 + k) * NGATE + n];
  g1c[idx] = acc;
}

// Pack fp32 weights into the WMMA B-fragment swizzle (bf16).
// K row k: k < kLim0 -> S0[k][n]; k >= kStart1 -> S1[k-kStart1][n]; else 0.
__global__ void hd_swizzle(const float* __restrict__ S0, const float* __restrict__ S1,
                           unsigned short* __restrict__ dst,
                           int nK, int nTiles, int N, int kLim0, int kStart1) {
  int idx = blockIdx.x * blockDim.x + threadIdx.x;       // one bf16 pair each
  int total = nTiles * nK * 32 * 8;
  if (idx >= total) return;
  int j    = idx & 7;
  int lane = (idx >> 3) & 31;
  int tk   = idx >> 8;
  int kt   = tk % nK;
  int ntile = tk / nK;
  int n = ntile * 16 + (lane & 15);
  int kbase = kt * 32 + (lane >> 4) * 8 + ((j < 4) ? 2 * j : 16 + 2 * (j - 4));  // B layout K pair
  unsigned int packed = 0;
#pragma unroll
  for (int e = 0; e < 2; ++e) {
    int k = kbase + e;
    float v = 0.f;
    if (k < kLim0)         v = S0[(size_t)k * N + n];
    else if (k >= kStart1) v = S1[(size_t)(k - kStart1) * N + n];
    packed |= ((unsigned int)f2bf(v)) << (16 * e);
  }
  ((unsigned int*)dst)[idx] = packed;
}

// =====================================================================
extern "C" void kernel_launch(void* const* d_in, const int* in_sizes, int n_in,
                              void* d_out, int out_size, void* d_ws, size_t ws_size,
                              hipStream_t stream) {
  const float* z   = (const float*)d_in[0];
  // d_in[1] pianoroll, d_in[2] ssm: unused by the reference computation
  const float* Wz  = (const float*)d_in[3];
  const float* bz  = (const float*)d_in[4];
  // d_in[5] cond_W1 unused (x0 == 0)
  const float* cU1 = (const float*)d_in[6];
  const float* cb1 = (const float*)d_in[7];
  const float* cW2 = (const float*)d_in[8];
  const float* cU2 = (const float*)d_in[9];
  const float* cb2 = (const float*)d_in[10];
  const float* Wcd = (const float*)d_in[11];
  const float* bcd = (const float*)d_in[12];
  const float* dW1 = (const float*)d_in[13];
  const float* dU1 = (const float*)d_in[14];
  const float* db1 = (const float*)d_in[15];
  const float* dW2 = (const float*)d_in[16];
  const float* dU2 = (const float*)d_in[17];
  const float* db2 = (const float*)d_in[18];
  const float* Wp  = (const float*)d_in[19];
  const float* bp  = (const float*)d_in[20];
  float* out = (float*)d_out;

  char* wsp = (char*)d_ws;
  size_t cur = 0;
  auto wsoff = [&](size_t bytes) -> void* {
    void* p = wsp + cur;
    cur += (bytes + 255) & ~(size_t)255;
    return p;
  };
  unsigned short* Wc1s = (unsigned short*)wsoff((size_t)704 * 2048 * 2);
  unsigned short* Wc2s = (unsigned short*)wsoff((size_t)1024 * 2048 * 2);
  unsigned short* Wps  = (unsigned short*)wsoff((size_t)512 * 176 * 2);
  float*          g1c  = (float*)wsoff((size_t)BATCH * NGATE * 4);
  unsigned short* dh1  = (unsigned short*)wsoff((size_t)BATCH * HID * 2);
  unsigned short* dh2  = (unsigned short*)wsoff((size_t)BATCH * HID * 2);
  float*          dc1  = (float*)wsoff((size_t)BATCH * HID * 4);
  float*          dc2  = (float*)wsoff((size_t)BATCH * HID * 4);
  float*          tmps = (float*)wsoff((size_t)BATCH * NGATE * 4);
  float*          h1c  = (float*)wsoff((size_t)BATCH * HID * 4);
  float*          ce   = (float*)wsoff((size_t)BATCH * HID * 4);

  const int TPB = 256;

  // weight swizzles (bf16 WMMA-B layout)
  {
    int p1 = 128 * NK1 * 32 * 8;
    hd_swizzle<<<(p1 + TPB - 1) / TPB, TPB, 0, stream>>>(dW1, dU1, Wc1s, NK1, 128,
                                                         NGATE, 176, 192);
    int p2 = 128 * NK2 * 32 * 8;
    hd_swizzle<<<(p2 + TPB - 1) / TPB, TPB, 0, stream>>>(dW2, dU2, Wc2s, NK2, 128,
                                                         NGATE, 512, 512);
    int p3 = NTP * NKP * 32 * 8;
    hd_swizzle<<<(p3 + TPB - 1) / TPB, TPB, 0, stream>>>(Wp, Wp, Wps, NKP, NTP,
                                                         NOUT, 512, 1 << 30);
  }

  // conductor prologue
  hd_dense_tanh<<<(BATCH * NGATE + TPB - 1) / TPB, TPB, 0, stream>>>(
      z, Wz, bz, tmps, 512, NGATE);
  hd_cond1<<<(BATCH * HID + TPB - 1) / TPB, TPB, 0, stream>>>(tmps, cU1, cb1, h1c);
  hd_cond2<<<(BATCH * HID + TPB - 1) / TPB, TPB, 0, stream>>>(tmps, h1c, cW2, cU2,
                                                              cb2, ce);
  hd_decinit<<<(BATCH * NGATE + TPB - 1) / TPB, TPB, 0, stream>>>(
      ce, Wcd, bcd, dh1, dc1, dh2, dc2);
  hd_g1const<<<(BATCH * NGATE + TPB - 1) / TPB, TPB, 0, stream>>>(ce, dW1, db1, g1c);

  // persistent 256-step decoder: 8 WGPs, 16 batch rows each
  hd_decoder_kernel<<<dim3(8), dim3(1024), 0, stream>>>(
      Wc1s, Wc2s, Wps, g1c, dh1, dh2, dc1, dc2, db2, bp, out);
}
